// NTM_63891933495587
// MI455X (gfx1250) — matrix operand order repaired and checked by reading, not compiled, not run on hardware
//
#include <hip/hip_runtime.h>

// ---------------------------------------------------------------------------
// NTM forward, persistent-kernel design for MI455X (gfx1250, wave32, WMMA).
// B=32,S=256,I=64,O=64,H=512,N=128,M=64.
// ---------------------------------------------------------------------------

typedef __bf16 bf16_t;
typedef __attribute__((ext_vector_type(16))) __bf16 v16bf;
typedef __attribute__((ext_vector_type(8)))  __bf16 v8bf;
typedef __attribute__((ext_vector_type(8)))  float  v8f;

#define kS   256
#define kH   512
#define kNS  128   // memory slots
#define kMM  64    // memory width
#define EPSF 1e-8f

// ---- dynamic LDS layout (bytes), per workgroup -----------------------------
#define SM_HBF    0        // bf16[16*512]  h (bf16)               16384
#define SM_XR     16384    // bf16[16*128]  [x_t ; r_{t-1}]         4096
#define SM_RBF    20480    // bf16[16*64]   r_t (bf16)              2048
#define SM_RF     22528    // f32 [16*64]   r_t (f32, atomically)   4096
#define SM_RAWR   26624    // f32 [16*80]   raw read head           5120
#define SM_RAWW   31744    // f32 [16*208]  raw write head         13312
#define SM_KTR    45056    // f32 [16*64]   tanh(key) read          4096
#define SM_KTW    49152    // f32 [16*64]   tanh(key) write         4096
#define SM_EL     53248    // f32 [16*64]   erase                   4096
#define SM_AL     57344    // f32 [16*64]   add                     4096
#define SM_SIMR   61440    // f32 [16*128]  sim / w_g scratch read  8192
#define SM_SIMW   69632    // f32 [16*128]  sim / w_g scratch write 8192
#define SM_WR     77824    // f32 [16*128]  w_r (persistent)        8192
#define SM_WW     86016    // f32 [16*128]  w_w (persistent)        8192
#define SM_SCAL   94208    // f32 [2*16*8]  beta,g,s0,s1,s2,gamma,invk 1024
#define SMEM_TOTAL 95232

// ---- workspace layout (bytes) ----------------------------------------------
#define WS_WXT   0u         // bf16 [2048][128]
#define WS_WHT   524288u    // bf16 [2048][512]
#define WS_WRT   2621440u   // bf16 [80][512]   (70 padded to 80)
#define WS_WWT   2703360u   // bf16 [208][512]  (198 padded to 208)
#define WS_WOHT  2916352u   // bf16 [64][512]
#define WS_WORT  2981888u   // bf16 [64][64]
#define WS_MEM   2990080u   // f32  [32][128][64]

// ---------------------------------------------------------------------------
__device__ __forceinline__ float sigmoidf_(float x) {
    return __builtin_amdgcn_rcpf(1.f + __expf(-x));
}
__device__ __forceinline__ float tanhf_(float x) {
    float cx = fminf(fmaxf(x, -15.f), 15.f);
    float e  = __expf(2.f * cx);
    return (e - 1.f) * __builtin_amdgcn_rcpf(e + 1.f);
}
__device__ __forceinline__ float softplusf_(float x) {
    return (x > 20.f) ? x : __logf(1.f + __expf(x));
}
__device__ __forceinline__ float wave_sum(float v) {
    #pragma unroll
    for (int m = 16; m >= 1; m >>= 1) v += __shfl_xor(v, m, 32);
    return v;
}
__device__ __forceinline__ float wave_max(float v) {
    #pragma unroll
    for (int m = 16; m >= 1; m >>= 1) v = fmaxf(v, __shfl_xor(v, m, 32));
    return v;
}
__device__ __forceinline__ v8f splat8(float v) {
    v8f r;
    #pragma unroll
    for (int i = 0; i < 8; ++i) r[i] = v;
    return r;
}
// Per-iteration VGPR barrier: rematerialize instead of spill/reload.
__device__ __forceinline__ float opaquef(float v) {
    asm volatile("" : "+v"(v));
    return v;
}

// A fragment (16x32 bf16, M x K), row-major in LDS with row stride lda.
// ISA 7.12.2: lanes 0-15 => M=lane, K = k0+{0..7,16..23}; lanes 16-31 => K+8.
__device__ __forceinline__ v16bf load_a_frag(const bf16_t* base, int lda, int k0, int lane) {
    int m   = lane & 15;
    int sel = (lane >> 4) << 3;            // 0 or 8
    const bf16_t* p = base + m * lda + k0 + sel;
    v8bf lo = *(const v8bf*)p;
    v8bf hi = *(const v8bf*)(p + 16);
    return __builtin_shufflevector(lo, hi, 0,1,2,3,4,5,6,7,8,9,10,11,12,13,14,15);
}
// B fragment (32x16 bf16, K x N) from transposed weights W^T [N][K] in global.
// lanes 0-15: N=lane, K=k0+{0..7,16..23} ; lanes 16-31: N=lane-16, K=k0+{8..15,24..31}.
// Explicit address_space(1) cast: the weight pointers are provenance-opaqued
// (anti-LICM), so force global_load_b128 rather than flat_load_b128 (flat
// would tie up DScnt and serialize against the LDS A-fragment loads).
__device__ __forceinline__ v16bf load_b_frag(const bf16_t* wt, int ldk, int n0, int k0, int lane) {
    const bf16_t* p = wt + (size_t)(n0 + (lane & 15)) * ldk + k0 + ((lane >> 4) << 3);
    typedef const v8bf __attribute__((address_space(1))) * gp_t;
    gp_t gp = (gp_t)p;
    v8bf lo = gp[0];
    v8bf hi = gp[2];
    return __builtin_shufflevector(lo, hi, 0,1,2,3,4,5,6,7,8,9,10,11,12,13,14,15);
}
__device__ __forceinline__ v8f wmma_bf16(v16bf a, v16bf b, v8f c) {
    return __builtin_amdgcn_wmma_f32_16x16x32_bf16(false, a, false, b, (short)0, c, false, false);
}

// ---------------------------------------------------------------------------
// Prologue: transpose + convert all weights to bf16 once per launch.
// ---------------------------------------------------------------------------
__global__ void k_prep_weights(const float* __restrict__ Wx, const float* __restrict__ Wh,
                               const float* __restrict__ Wr, const float* __restrict__ Ww,
                               const float* __restrict__ Wo,
                               bf16_t* wxT, bf16_t* whT, bf16_t* wrT,
                               bf16_t* wwT, bf16_t* wohT, bf16_t* worT) {
    long i = (long)blockIdx.x * 256 + threadIdx.x;
    if (i < 262144)  { long n = i / 128, k = i % 128; wxT[i] = (bf16_t)Wx[k * 2048 + n]; return; }
    i -= 262144;
    if (i < 1048576) { long n = i / 512, k = i % 512; whT[i] = (bf16_t)Wh[k * 2048 + n]; return; }
    i -= 1048576;
    if (i < 40960)   { long n = i / 512, k = i % 512; wrT[i] = (n < 70)  ? (bf16_t)Wr[k * 70 + n]  : (bf16_t)0.f; return; }
    i -= 40960;
    if (i < 106496)  { long n = i / 512, k = i % 512; wwT[i] = (n < 198) ? (bf16_t)Ww[k * 198 + n] : (bf16_t)0.f; return; }
    i -= 106496;
    if (i < 32768)   { long n = i / 512, k = i % 512; wohT[i] = (bf16_t)Wo[k * 64 + n]; return; }
    i -= 32768;
    if (i < 4096)    { long n = i / 64,  k = i % 64;  worT[i] = (bf16_t)Wo[(512 + k) * 64 + n]; return; }
}

__global__ void k_init_mem(float* mem) {
    int i = blockIdx.x * 256 + threadIdx.x;
    if (i < 32 * kNS * kMM) mem[i] = 1e-6f;
}

// ---------------------------------------------------------------------------
// Persistent NTM kernel: one workgroup per 16-batch group, 32 waves.
// ---------------------------------------------------------------------------
__global__ __launch_bounds__(1024)
void ntm_persistent(const float* __restrict__ x,
                    const float* __restrict__ lstm_b,
                    const float* __restrict__ b_read,
                    const float* __restrict__ b_write,
                    const float* __restrict__ b_out,
                    const bf16_t* __restrict__ WxT0, const bf16_t* __restrict__ WhT0,
                    const bf16_t* __restrict__ WrT0, const bf16_t* __restrict__ WwT0,
                    const bf16_t* __restrict__ WohT0, const bf16_t* __restrict__ WorT0,
                    float* __restrict__ mem_all,
                    float* __restrict__ out) {
    extern __shared__ char smem[];
    bf16_t* h_bf  = (bf16_t*)(smem + SM_HBF);
    bf16_t* xr_s  = (bf16_t*)(smem + SM_XR);
    bf16_t* rbf_s = (bf16_t*)(smem + SM_RBF);
    float*  r_f   = (float*)(smem + SM_RF);
    float*  raw_r = (float*)(smem + SM_RAWR);
    float*  raw_w = (float*)(smem + SM_RAWW);
    float*  ktr_s = (float*)(smem + SM_KTR);
    float*  ktw_s = (float*)(smem + SM_KTW);
    float*  e_s   = (float*)(smem + SM_EL);
    float*  a_s   = (float*)(smem + SM_AL);
    float*  simr  = (float*)(smem + SM_SIMR);
    float*  simw  = (float*)(smem + SM_SIMW);
    float*  w_r_s = (float*)(smem + SM_WR);
    float*  w_w_s = (float*)(smem + SM_WW);
    float*  scal  = (float*)(smem + SM_SCAL);

    const int tid  = threadIdx.x;
    const int lane = tid & 31;
    const int wv   = tid >> 5;           // wave id 0..31
    const int grp  = blockIdx.x;         // batch group (16 each)
    float* memg = mem_all + (size_t)grp * 16 * kNS * kMM;

    // Loop-carried opaque copies of the weight pointers: prevents LICM from
    // hoisting the (time-invariant) B-fragment loads out of the t-loop, which
    // previously forced a mass spill of v16bf fragments to scratch.
    const bf16_t* WxT  = WxT0;
    const bf16_t* WhT  = WhT0;
    const bf16_t* WrT  = WrT0;
    const bf16_t* WwT  = WwT0;
    const bf16_t* WohT = WohT0;
    const bf16_t* WorT = WorT0;

    // ---- per-wave persistent state ----
    const int hc   = wv * 16;            // 16 h-columns owned by this wave
    const int colg = hc + (lane & 15);
    const float bi_ = lstm_b[colg];
    const float bf_ = lstm_b[512 + colg];
    const float bg_ = lstm_b[1024 + colg];
    const float bo_ = lstm_b[1536 + colg];
    float c_reg[8];
    #pragma unroll
    for (int e = 0; e < 8; ++e) c_reg[e] = 0.f;

    // ---- state init ----
    for (int idx = tid; idx < 16 * kH; idx += 1024) h_bf[idx] = (bf16_t)0.f;
    for (int idx = tid; idx < 16 * kNS; idx += 1024) {
        float v = ((idx & (kNS - 1)) == 0) ? 1.f : 0.f;
        w_r_s[idx] = v; w_w_s[idx] = v;
    }
    { r_f[tid & 1023] = 0.f; }   // 16*64 == 1024
    __syncthreads();

    for (int t = 0; t < kS; ++t) {
        // Defeat loop-invariant hoisting of all weight loads (see above).
        asm volatile("" : "+s"(WxT), "+s"(WhT), "+s"(WrT),
                          "+s"(WwT), "+s"(WohT), "+s"(WorT));

        // ===== A: build [x_t ; r_{t-1}] bf16 =====
        for (int idx = tid; idx < 16 * 128; idx += 1024) {
            int b = idx >> 7, j = idx & 127;
            float v = (j < 64) ? x[(((size_t)(grp * 16 + b)) * kS + t) * 64 + j]
                               : r_f[b * 64 + (j - 64)];
            xr_s[idx] = (bf16_t)v;
        }
        __syncthreads();

        // ===== B1: gates = [x;r]@Wx + h@Wh + b   (4 tiles per wave) =====
        // opaquef(): rematerialize bias splats per step instead of spilling
        // four live v8f vectors across the whole loop body.
        v8f ai = splat8(opaquef(bi_));
        v8f af = splat8(opaquef(bf_));
        v8f ag = splat8(opaquef(bg_));
        v8f ao = splat8(opaquef(bo_));
        #pragma unroll
        for (int kt = 0; kt < 4; ++kt) {
            v16bf a = load_a_frag(xr_s, 128, kt * 32, lane);
            ai = wmma_bf16(a, load_b_frag(WxT, 128, hc,        kt * 32, lane), ai);
            af = wmma_bf16(a, load_b_frag(WxT, 128, 512 + hc,  kt * 32, lane), af);
            ag = wmma_bf16(a, load_b_frag(WxT, 128, 1024 + hc, kt * 32, lane), ag);
            ao = wmma_bf16(a, load_b_frag(WxT, 128, 1536 + hc, kt * 32, lane), ao);
        }
        #pragma unroll 4
        for (int kt = 0; kt < 16; ++kt) {
            v16bf a = load_a_frag(h_bf, kH, kt * 32, lane);
            ai = wmma_bf16(a, load_b_frag(WhT, kH, hc,        kt * 32, lane), ai);
            af = wmma_bf16(a, load_b_frag(WhT, kH, 512 + hc,  kt * 32, lane), af);
            ag = wmma_bf16(a, load_b_frag(WhT, kH, 1024 + hc, kt * 32, lane), ag);
            ao = wmma_bf16(a, load_b_frag(WhT, kH, 1536 + hc, kt * 32, lane), ao);
        }
        __syncthreads();   // everyone done reading old h

        // ===== B2: LSTM nonlinearity, c stays in registers, h -> LDS bf16 =====
        {
            int mrow = (lane >> 4) << 3;
            #pragma unroll
            for (int e = 0; e < 8; ++e) {
                float gi = sigmoidf_(ai[e]);
                float gf = sigmoidf_(af[e]);
                float gg = tanhf_(ag[e]);
                float go = sigmoidf_(ao[e]);
                float cc = gf * c_reg[e] + gi * gg;
                c_reg[e] = cc;
                h_bf[(mrow + e) * kH + colg] = (bf16_t)(go * tanhf_(cc));
            }
        }
        __syncthreads();

        // ===== C: head GEMMs  raw = h @ W_{read,write} + b  (18 tiles) =====
        if (wv < 18) {
            bool rd = (wv < 5);
            int n0 = rd ? 16 * wv : 16 * (wv - 5);
            const bf16_t* WT = rd ? WrT : WwT;
            const float* bias = rd ? b_read : b_write;
            int valid  = rd ? 70 : 198;
            int stride = rd ? 80 : 208;
            float* raw = rd ? raw_r : raw_w;
            int col = n0 + (lane & 15);
            v8f acc = splat8(opaquef((col < valid) ? bias[col] : 0.f));
            #pragma unroll 4
            for (int kt = 0; kt < 16; ++kt) {
                v16bf a = load_a_frag(h_bf, kH, kt * 32, lane);
                acc = wmma_bf16(a, load_b_frag(WT, kH, n0, kt * 32, lane), acc);
            }
            int mrow = (lane >> 4) << 3;
            #pragma unroll
            for (int e = 0; e < 8; ++e) raw[(mrow + e) * stride + col] = acc[e];
        }
        __syncthreads();

        // ===== D1: keys/erase/add transforms + per-head scalars =====
        {
            int b = tid >> 6, j = tid & 63;
            ktr_s[tid] = tanhf_(raw_r[b * 80 + j]);
            ktw_s[tid] = tanhf_(raw_w[b * 208 + j]);
            e_s[tid]   = sigmoidf_(raw_w[b * 208 + 70 + j]);
            a_s[tid]   = tanhf_(raw_w[b * 208 + 134 + j]);
            r_f[tid]   = 0.f;
            if (tid < 32) {
                int b2 = tid & 15, hd = tid >> 4;
                const float* p = (hd ? raw_w + b2 * 208 : raw_r + b2 * 80) + 64;
                float mx = fmaxf(p[2], fmaxf(p[3], p[4]));
                float s0 = __expf(p[2] - mx), s1 = __expf(p[3] - mx), s2 = __expf(p[4] - mx);
                float si = __builtin_amdgcn_rcpf(s0 + s1 + s2);
                float* sc = scal + (hd * 16 + b2) * 8;
                sc[0] = softplusf_(p[0]);           // beta
                sc[1] = sigmoidf_(p[1]);            // g
                sc[2] = s0 * si; sc[3] = s1 * si; sc[4] = s2 * si;
                sc[5] = 1.f + softplusf_(p[5]);     // gamma
            }
        }
        __syncthreads();
        // ===== D2: key inverse norms =====
        if (tid < 32) {
            int b2 = tid & 15, hd = tid >> 4;
            const float* kt_ = (hd ? ktw_s : ktr_s) + b2 * 64;
            float s = 0.f;
            for (int j = 0; j < 64; ++j) { float v = kt_[j]; s += v * v; }
            scal[(hd * 16 + b2) * 8 + 6] = __builtin_amdgcn_rcpf(__fsqrt_rn(s) + EPSF);
        }
        __syncthreads();

        // ===== E: fused cosine similarity for BOTH heads (one pass over mem).
        // Half-wave scheme: 16 lanes x float4 cover one 64-wide slot row
        // (coalesced b128); 4-step butterfly reduce; 2 slots per iteration. =====
        {
            int b = wv >> 1, nb = (wv & 1) << 6;
            int half = lane >> 4;            // 0/1
            int j0 = (lane & 15) << 2;       // 0..60
            float4 kr4 = *(const float4*)(ktr_s + b * 64 + j0);
            float4 kw4 = *(const float4*)(ktw_s + b * 64 + j0);
            float ivr = scal[b * 8 + 6], ivw = scal[(16 + b) * 8 + 6];
            const float* mp = memg + (size_t)(b * kNS + nb + half) * kMM + j0;
            for (int i = 0; i < 32; ++i) {
                float4 m = *(const float4*)mp;
                float dr = m.x * kr4.x + m.y * kr4.y + m.z * kr4.z + m.w * kr4.w;
                float dw = m.x * kw4.x + m.y * kw4.y + m.z * kw4.z + m.w * kw4.w;
                float s2 = m.x * m.x + m.y * m.y + m.z * m.z + m.w * m.w;
                #pragma unroll
                for (int msk = 8; msk >= 1; msk >>= 1) {
                    dr += __shfl_xor(dr, msk, 32);
                    dw += __shfl_xor(dw, msk, 32);
                    s2 += __shfl_xor(s2, msk, 32);
                }
                if ((lane & 15) == 0) {
                    int n = nb + 2 * i + half;
                    float inm = __builtin_amdgcn_rcpf(__fsqrt_rn(s2) + EPSF);
                    simr[b * kNS + n] = dr * ivr * inm;
                    simw[b * kNS + n] = dw * ivw * inm;
                }
                mp += 2 * kMM;
            }
        }
        __syncthreads();

        // ===== F: addressing (softmax, interpolate, shift, sharpen); wave per (head,b) =====
        {
            int hd = wv >> 4, b = wv & 15;
            float* S_ = hd ? simw : simr;
            float* wp = hd ? w_w_s : w_r_s;
            const float* sc = scal + (hd * 16 + b) * 8;
            float beta = sc[0], g = sc[1], s0 = sc[2], s1 = sc[3], s2 = sc[4], gamma = sc[5];
            float vt[4];
            #pragma unroll
            for (int q = 0; q < 4; ++q) vt[q] = beta * S_[b * kNS + lane + 32 * q];
            float mx = fmaxf(fmaxf(vt[0], vt[1]), fmaxf(vt[2], vt[3]));
            mx = wave_max(mx);
            float ex[4], ssum = 0.f;
            #pragma unroll
            for (int q = 0; q < 4; ++q) { ex[q] = __expf(vt[q] - mx); ssum += ex[q]; }
            ssum = wave_sum(ssum);
            float rin = __builtin_amdgcn_rcpf(ssum);
            #pragma unroll
            for (int q = 0; q < 4; ++q) {
                int n = lane + 32 * q;
                float wc = ex[q] * rin;
                S_[b * kNS + n] = g * wc + (1.f - g) * wp[b * kNS + n];   // w_g scratch
            }
            asm volatile("s_wait_dscnt 0" ::: "memory");
            float wt[4], tot = 0.f;
            #pragma unroll
            for (int q = 0; q < 4; ++q) {
                int n = lane + 32 * q;
                float wg = s0 * S_[b * kNS + ((n + 1) & 127)]
                         + s1 * S_[b * kNS + n]
                         + s2 * S_[b * kNS + ((n - 1) & 127)];
                wt[q] = __expf(gamma * __logf(wg + EPSF));                // sharpen
                tot += wt[q];
            }
            tot = wave_sum(tot);
            float ti = __builtin_amdgcn_rcpf(tot);
            #pragma unroll
            for (int q = 0; q < 4; ++q) wp[b * kNS + lane + 32 * q] = wt[q] * ti;
        }
        __syncthreads();

        // ===== G: fused read-vector + erase/add memory update (one RMW pass),
        // half-wave x float4, coalesced b128 load+store =====
        {
            int b = wv >> 1, nb = (wv & 1) << 6;
            int half = lane >> 4;
            int j0 = (lane & 15) << 2;
            float4 e4 = *(const float4*)(e_s + b * 64 + j0);
            float4 a4 = *(const float4*)(a_s + b * 64 + j0);
            float* mp = memg + (size_t)(b * kNS + nb + half) * kMM + j0;
            float4 racc = make_float4(0.f, 0.f, 0.f, 0.f);
            for (int i = 0; i < 32; ++i) {
                int n = nb + 2 * i + half;
                float wr = w_r_s[b * kNS + n];
                float ww = w_w_s[b * kNS + n];
                float4 m = *(float4*)mp;
                racc.x += wr * m.x; racc.y += wr * m.y;
                racc.z += wr * m.z; racc.w += wr * m.w;
                m.x = m.x * (1.f - ww * e4.x) + ww * a4.x;
                m.y = m.y * (1.f - ww * e4.y) + ww * a4.y;
                m.z = m.z * (1.f - ww * e4.z) + ww * a4.z;
                m.w = m.w * (1.f - ww * e4.w) + ww * a4.w;
                *(float4*)mp = m;
                mp += 2 * kMM;
            }
            atomicAdd(&r_f[b * 64 + j0 + 0], racc.x);
            atomicAdd(&r_f[b * 64 + j0 + 1], racc.y);
            atomicAdd(&r_f[b * 64 + j0 + 2], racc.z);
            atomicAdd(&r_f[b * 64 + j0 + 3], racc.w);
        }
        __syncthreads();

        // ===== H: r -> bf16 =====
        rbf_s[tid & 1023] = (bf16_t)r_f[tid & 1023];
        __syncthreads();

        // ===== I: out = [h;r] @ W_out + b_out  (4 tiles, waves 0-3) =====
        if (wv < 4) {
            int n0 = wv * 16;
            int col = n0 + (lane & 15);
            v8f acc = splat8(opaquef(b_out[col]));
            #pragma unroll 4
            for (int kt = 0; kt < 16; ++kt) {
                v16bf a = load_a_frag(h_bf, kH, kt * 32, lane);
                acc = wmma_bf16(a, load_b_frag(WohT, kH, n0, kt * 32, lane), acc);
            }
            #pragma unroll
            for (int kt = 0; kt < 2; ++kt) {
                v16bf a = load_a_frag(rbf_s, 64, kt * 32, lane);
                acc = wmma_bf16(a, load_b_frag(WorT, 64, n0, kt * 32, lane), acc);
            }
            int mrow = (lane >> 4) << 3;
            #pragma unroll
            for (int e = 0; e < 8; ++e)
                out[(((size_t)(grp * 16 + mrow + e)) * kS + t) * 64 + col] = acc[e];
        }
        // next iteration's phase-A barrier orders I(t) vs B(t+1)
    }
}

// ---------------------------------------------------------------------------
extern "C" void kernel_launch(void* const* d_in, const int* in_sizes, int n_in,
                              void* d_out, int out_size, void* d_ws, size_t ws_size,
                              hipStream_t stream) {
    const float* x       = (const float*)d_in[0];
    const float* lstm_Wx = (const float*)d_in[1];
    const float* lstm_Wh = (const float*)d_in[2];
    const float* lstm_b  = (const float*)d_in[3];
    const float* W_read  = (const float*)d_in[4];
    const float* b_read  = (const float*)d_in[5];
    const float* W_write = (const float*)d_in[6];
    const float* b_write = (const float*)d_in[7];
    const float* W_out   = (const float*)d_in[8];
    const float* b_out   = (const float*)d_in[9];

    char* ws = (char*)d_ws;
    bf16_t* WxT  = (bf16_t*)(ws + WS_WXT);
    bf16_t* WhT  = (bf16_t*)(ws + WS_WHT);
    bf16_t* WrT  = (bf16_t*)(ws + WS_WRT);
    bf16_t* WwT  = (bf16_t*)(ws + WS_WWT);
    bf16_t* WohT = (bf16_t*)(ws + WS_WOHT);
    bf16_t* WorT = (bf16_t*)(ws + WS_WORT);
    float*  memg = (float*)(ws + WS_MEM);

    const long totalW = 262144 + 1048576 + 40960 + 106496 + 32768 + 4096;
    k_prep_weights<<<(int)((totalW + 255) / 256), 256, 0, stream>>>(
        lstm_Wx, lstm_Wh, W_read, W_write, W_out, WxT, WhT, WrT, WwT, WohT, WorT);
    k_init_mem<<<(32 * kNS * kMM + 255) / 256, 256, 0, stream>>>(memg);

    ntm_persistent<<<2, 1024, SMEM_TOTAL, stream>>>(
        x, lstm_b, b_read, b_write, b_out,
        WxT, WhT, WrT, WwT, WohT, WorT,
        memg, (float*)d_out);
}
